// Head_17806934410015
// MI455X (gfx1250) — matrix operand order repaired and checked by reading, not compiled, not run on hardware
//
#include <hip/hip_runtime.h>
#include <math.h>

typedef _Float16 h16 __attribute__((ext_vector_type(16)));
typedef _Float16 h8  __attribute__((ext_vector_type(8)));
typedef float    f8  __attribute__((ext_vector_type(8)));
typedef float    f4  __attribute__((ext_vector_type(4)));

#define WMMA16x16x32(a,b,c) \
  __builtin_amdgcn_wmma_f32_16x16x32_f16(false,(a),false,(b),(short)0,(c),false,false)
#define SCHED_FENCE() __builtin_amdgcn_sched_barrier(0)

static constexpr int B  = 8;
static constexpr int T  = 2048;
static constexpr int D  = 1024;
static constexpr int HS = 128;

union AU { h16 v; h8 h[2]; _Float16 e[16]; };

// ---------------- Kernel 0: convert weights to f16, fold softmax scale into Wq
__global__ __launch_bounds__(256) void cvt_w(const float* __restrict__ Wk,
                                             const float* __restrict__ Wq,
                                             const float* __restrict__ Wv,
                                             _Float16* __restrict__ Wh) {
  int i = blockIdx.x * 256 + threadIdx.x;
  if (i < HS * D) {
    Wh[i]          = (_Float16)Wk[i];
    Wh[HS*D + i]   = (_Float16)(Wq[i] * 0.08838834764831845f); // 1/sqrt(128)
    Wh[2*HS*D + i] = (_Float16)Wv[i];
  }
}

// ---------------- Kernel 1: QKV projection via WMMA, ping-pong pipelined
// one wave computes a 16(t) x 128(h) tile of one of {K, Q, V}; K-dim = D = 1024
__global__ __launch_bounds__(256) void proj(const float* __restrict__ x,
                                            const _Float16* __restrict__ Wh,
                                            _Float16* __restrict__ kh,
                                            _Float16* __restrict__ qh,
                                            _Float16* __restrict__ vT) {
  int wave = (blockIdx.x * blockDim.x + threadIdx.x) >> 5;
  int lane = threadIdx.x & 31;
  int m   = wave / (B * (T/16));          // 0=K, 1=Q, 2=V
  int rem = wave % (B * (T/16));
  int b   = rem / (T/16);
  int tq  = rem % (T/16);
  int qi  = lane & 15;
  int hi  = lane >> 4;

  const float*    xa = x  + (size_t)(b*T + tq*16 + qi) * D;   // A: lane = row t
  const _Float16* W  = Wh + (size_t)m * HS * D;

  f8  acc[8] = {};
  f4  u[2][4];          // double-buffered raw A (f32)
  h16 bbuf[2][8];       // double-buffered B fragments

  // prologue: issue chunk 0 loads
  {
    const f4* p0 = (const f4*)(xa + hi*16);
    const f4* p1 = (const f4*)(xa + hi*16 + 8);
    u[0][0] = p0[0]; u[0][1] = p0[1]; u[0][2] = p1[0]; u[0][3] = p1[1];
    #pragma unroll
    for (int nt = 0; nt < 8; nt++)
      bbuf[0][nt] = *(const h16*)(W + (size_t)(nt*16 + qi) * D + hi*16);
  }

  #pragma unroll 2
  for (int k0 = 0; k0 < D; k0 += 32) {
    int s = (k0 >> 5) & 1;
    // ---- issue next chunk's loads first (hidden behind this chunk's WMMAs) ----
    if (k0 + 32 < D) {
      int kn = k0 + 32, sn = s ^ 1;
      const f4* p0 = (const f4*)(xa + kn + hi*16);
      const f4* p1 = (const f4*)(xa + kn + hi*16 + 8);
      u[sn][0] = p0[0]; u[sn][1] = p0[1]; u[sn][2] = p1[0]; u[sn][3] = p1[1];
      #pragma unroll
      for (int nt = 0; nt < 8; nt++)
        bbuf[sn][nt] = *(const h16*)(W + (size_t)(nt*16 + qi) * D + kn + hi*16);
    }
    SCHED_FENCE();
    // ---- convert A fragment (data requested one chunk ago) ----
    AU a;
    #pragma unroll
    for (int e = 0; e < 4; e++) {
      a.e[e]      = (_Float16)u[s][0][e];
      a.e[4 + e]  = (_Float16)u[s][1][e];
      a.e[8 + e]  = (_Float16)u[s][2][e];
      a.e[12 + e] = (_Float16)u[s][3][e];
    }
    #pragma unroll
    for (int nt = 0; nt < 8; nt++)
      acc[nt] = WMMA16x16x32(a.v, bbuf[s][nt], acc[nt]);
  }

  // C layout: reg r -> row t = tq*16 + r + 8*hi ; lane -> col h = nt*16 + qi
  if (m == 2) {
    // vT[b][h][t]: per lane the 8 regs are contiguous in t -> one b128 store each
    #pragma unroll
    for (int nt = 0; nt < 8; nt++) {
      int h = nt*16 + qi;
      h8 pk;
      #pragma unroll
      for (int r = 0; r < 8; r++) pk[r] = (_Float16)acc[nt][r];
      *(h8*)(vT + (size_t)(b*HS + h) * T + tq*16 + 8*hi) = pk;
    }
  } else {
    _Float16* out = (m == 0) ? kh : qh;
    #pragma unroll
    for (int nt = 0; nt < 8; nt++)
      #pragma unroll
      for (int r = 0; r < 8; r++) {
        int h = nt*16 + qi;
        int t = tq*16 + r + 8*hi;
        out[(size_t)(b*T + t) * HS + h] = (_Float16)acc[nt][r];
      }
  }
}

// ---------------- Kernel 2: causal flash attention, one wave per 16-query tile
// S^T = K_tile * Q^T   (C layout: lane = query, regs = key)
// O^T = V^T * P^T      (C layout: lane = query, regs = head-dim)
__global__ __launch_bounds__(256) void attn(const _Float16* __restrict__ qh,
                                            const _Float16* __restrict__ kh,
                                            const _Float16* __restrict__ vT,
                                            float* __restrict__ out) {
  int wave = (blockIdx.x * blockDim.x + threadIdx.x) >> 5;
  int lane = threadIdx.x & 31;
  int b  = wave / (T/16);
  int tq = wave % (T/16);
  int t0 = tq * 16;
  int qi = lane & 15;
  int hi = lane >> 4;

  // Q as B fragments: lane = query col, K(head) = 32c + hi*16 + 0..15 contiguous
  h16 qf[4];
  const _Float16* qrow = qh + (size_t)(b*T + t0 + qi) * HS;
  #pragma unroll
  for (int c = 0; c < 4; c++)
    qf[c] = *(const h16*)(qrow + 32*c + hi*16);

  float mrun = -INFINITY, lrun = 0.f;
  f8 o[8] = {};

  const _Float16* kbase = kh + (size_t)b * T * HS;
  const _Float16* vbase = vT + (size_t)b * HS * T;

  for (int kb = 0; kb <= t0; kb += 32) {
    bool act1 = (kb + 16 <= t0);

    // ---- subtile 0: batch 8 loads, fence, 4 WMMAs ----
    AU kf0[4];
    const _Float16* krow0 = kbase + (size_t)(kb + qi) * HS;
    #pragma unroll
    for (int c = 0; c < 4; c++) {
      kf0[c].h[0] = *(const h8*)(krow0 + 32*c + hi*16);
      kf0[c].h[1] = *(const h8*)(krow0 + 32*c + hi*16 + 8);
    }
    SCHED_FENCE();
    f8 s0 = {};
    #pragma unroll
    for (int c = 0; c < 4; c++) s0 = WMMA16x16x32(kf0[c].v, qf[c], s0);

    // ---- subtile 1 ----
    f8 s1;
    if (act1) {
      AU kf1[4];
      const _Float16* krow1 = kbase + (size_t)(kb + 16 + qi) * HS;
      #pragma unroll
      for (int c = 0; c < 4; c++) {
        kf1[c].h[0] = *(const h8*)(krow1 + 32*c + hi*16);
        kf1[c].h[1] = *(const h8*)(krow1 + 32*c + hi*16 + 8);
      }
      SCHED_FENCE();
      f8 t = {};
      #pragma unroll
      for (int c = 0; c < 4; c++) t = WMMA16x16x32(kf1[c].v, qf[c], t);
      s1 = t;
    }

    // ---- issue ALL V^T fragment loads now; softmax below hides their latency ----
    AU av[8];
    #pragma unroll
    for (int t = 0; t < 8; t++) {
      const _Float16* vrow = vbase + (size_t)(t*16 + qi) * T + kb; // lane = hd row
      av[t].h[0] = *(const h8*)(vrow + hi*16);
      av[t].h[1] = *(const h8*)(vrow + hi*16 + 8);
    }
    SCHED_FENCE();

    // causal masking (diagonal subtile / fully-masked subtile)
    if (kb == t0) {
      #pragma unroll
      for (int r = 0; r < 8; r++) if ((r + 8*hi) > qi) s0[r] = -INFINITY;
    }
    if (act1) {
      if (kb + 16 == t0) {
        #pragma unroll
        for (int r = 0; r < 8; r++) if ((r + 8*hi) > qi) s1[r] = -INFINITY;
      }
    } else {
      #pragma unroll
      for (int r = 0; r < 8; r++) s1[r] = -INFINITY;
    }

    // ---- online softmax (per-query state per lane; pair = lane, lane^16) ----
    float bm = -INFINITY;
    #pragma unroll
    for (int r = 0; r < 8; r++) bm = fmaxf(bm, fmaxf(s0[r], s1[r]));
    bm = fmaxf(bm, __shfl_xor(bm, 16, 32));
    float mnew  = fmaxf(mrun, bm);
    float alpha = __expf(mrun - mnew);

    float p0[8], p1[8], ps = 0.f;
    #pragma unroll
    for (int r = 0; r < 8; r++) {
      p0[r] = __expf(s0[r] - mnew);
      p1[r] = __expf(s1[r] - mnew);
      ps += p0[r] + p1[r];
    }
    lrun = lrun * alpha + ps + __shfl_xor(ps, 16, 32);

    // pack P into B-fragment layout (lane = query, K = hi*16 + e)
    h16 pb;
    #pragma unroll
    for (int r = 0; r < 8; r++) {
      float y0 = __shfl_xor(p0[r], 16, 32);
      float y1 = __shfl_xor(p1[r], 16, 32);
      pb[r]     = (_Float16)(hi ? y1 : p0[r]);   // keys kb + hi*16 + r
      pb[8 + r] = (_Float16)(hi ? p1[r] : y0);   // keys kb + hi*16 + 8 + r
    }

    // ---- rescale + PV WMMA chain (V data already resident) ----
    #pragma unroll
    for (int t = 0; t < 8; t++) {
      #pragma unroll
      for (int r = 0; r < 8; r++) o[t][r] *= alpha;
      o[t] = WMMA16x16x32(av[t].v, pb, o[t]);
    }
    mrun = mnew;
  }

  float rl = 1.f / lrun;
  #pragma unroll
  for (int t = 0; t < 8; t++)
    #pragma unroll
    for (int r = 0; r < 8; r++) {
      int h = t*16 + r + 8*hi;
      out[(size_t)(b*T + t0 + qi) * HS + h] = o[t][r] * rl;
    }
}

extern "C" void kernel_launch(void* const* d_in, const int* in_sizes, int n_in,
                              void* d_out, int out_size, void* d_ws, size_t ws_size,
                              hipStream_t stream) {
  const float* x  = (const float*)d_in[0];
  const float* Wk = (const float*)d_in[1];
  const float* Wq = (const float*)d_in[2];
  const float* Wv = (const float*)d_in[3];
  float* out = (float*)d_out;

  _Float16* ws = (_Float16*)d_ws;
  _Float16* Wh = ws;                               // 3*HS*D halves
  _Float16* kh = Wh + (size_t)3*HS*D;              // B*T*HS
  _Float16* qh = kh + (size_t)B*T*HS;              // B*T*HS
  _Float16* vT = qh + (size_t)B*T*HS;              // B*HS*T (transposed V)

  cvt_w<<<(HS*D + 255)/256, 256, 0, stream>>>(Wk, Wq, Wv, Wh);
  proj <<<(3*B*(T/16))/8,   256, 0, stream>>>(x, Wh, kh, qh, vT);
  attn <<<(B*(T/16))/8,     256, 0, stream>>>(qh, kh, vT, out);
}
